// CPRLinearMixedFused_64330020159894
// MI455X (gfx1250) — matrix operand order repaired and checked by reading, not compiled, not run on hardware
//
#include <hip/hip_runtime.h>
#include <hip/hip_fp16.h>

typedef __attribute__((ext_vector_type(16))) _Float16 v16h;
typedef __attribute__((ext_vector_type(4)))  _Float16 v4h;
typedef __attribute__((ext_vector_type(8)))  float    v8f;

union Frag16 { uint4 u[2]; v16h h; };

constexpr int TM  = 128;   // workgroup tile M
constexpr int TN  = 128;   // workgroup tile N
constexpr int TK  = 64;    // K per stage (2 wmma k-steps of 32)
constexpr int LDA = 72;    // padded f16 row stride (144B = 36 dwords, conflict-free)
constexpr int LDB = 72;

// ---------------------------------------------------------------------------
// Pass 1: invert the K permutation.  inv[col_indices[k]] = k
// ---------------------------------------------------------------------------
__global__ void k_invert(const int* __restrict__ ci, int* __restrict__ inv, int K) {
    int k = blockIdx.x * blockDim.x + threadIdx.x;
    if (k < K) inv[ci[k]] = k;
}

// ---------------------------------------------------------------------------
// Pass 2: x f32 -> f16 (row-major), so GEMM tiles can be DMA'd LDS-direct.
// ---------------------------------------------------------------------------
__global__ __launch_bounds__(256) void k_convert(const float* __restrict__ X,
                                                 _Float16* __restrict__ Xh,
                                                 size_t n) {
    size_t i = ((size_t)blockIdx.x * 256 + threadIdx.x) * 8;
    if (i >= n) return;
    float4 a = *(const float4*)(X + i);
    float4 b = *(const float4*)(X + i + 4);
    union { v4h h[2]; uint4 u; } p;
    p.h[0].x = (_Float16)a.x; p.h[0].y = (_Float16)a.y;
    p.h[0].z = (_Float16)a.z; p.h[0].w = (_Float16)a.w;
    p.h[1].x = (_Float16)b.x; p.h[1].y = (_Float16)b.y;
    p.h[1].z = (_Float16)b.z; p.h[1].w = (_Float16)b.w;
    *(uint4*)(Xh + i) = p.u;
}

// ---------------------------------------------------------------------------
// Pass 3: build K-major dequantized f16 weight:
//   Bt[n*K + p] = f16( W[inv[p]*N + n] * scales[(inv[p]/GS)*N + n] )
// ---------------------------------------------------------------------------
__global__ __launch_bounds__(256) void k_dequant(const int* __restrict__ W,
                                                 const float* __restrict__ scales,
                                                 const int* __restrict__ inv,
                                                 _Float16* __restrict__ Bt,
                                                 int K, int N, int GS) {
    __shared__ _Float16 tile[64][65];
    const int j0 = blockIdx.y * 64;
    const int n0 = blockIdx.x * 64;
    const int t  = threadIdx.x;
#pragma unroll
    for (int i = 0; i < 16; ++i) {
        int e  = i * 256 + t;
        int jj = e >> 6, nn = e & 63;          // n fastest -> coalesced reads
        int src = inv[j0 + jj];
        float w = (float)W[(size_t)src * N + n0 + nn];
        float s = scales[(size_t)(src / GS) * N + n0 + nn];
        tile[jj][nn] = (_Float16)(w * s);
    }
    __syncthreads();
#pragma unroll
    for (int i = 0; i < 16; ++i) {
        int e  = i * 256 + t;
        int nn = e >> 6, jj = e & 63;          // k fastest -> coalesced writes
        Bt[(size_t)(n0 + nn) * K + j0 + jj] = tile[jj][nn];
    }
}

// ---------------------------------------------------------------------------
// Pass 4: f16 WMMA GEMM with async global->LDS double buffering.
//   256 threads = 8 waves (wave32).  Wave grid 2(M) x 4(N); each wave owns a
//   64x32 C tile = 4x2 fragments of v_wmma_f32_16x16x32_f16.
// ---------------------------------------------------------------------------
__global__ __launch_bounds__(256) void k_gemm(const _Float16* __restrict__ Xh,
                                              const _Float16* __restrict__ Bt,
                                              const float* __restrict__ bias,
                                              float* __restrict__ Y,
                                              int M, int N, int K) {
    __shared__ _Float16 As[2][TM * LDA];
    __shared__ _Float16 Bs[2][TN * LDB];

    const int t     = threadIdx.x;
    const int lane  = t & 31;
    const int wave  = t >> 5;
    const int waveM = wave >> 2;     // 0..1  -> 64-row slab
    const int waveN = wave & 3;      // 0..3  -> 32-col slab
    const int half  = lane >> 4;     // lane half selects K sub-range
    const int l     = lane & 15;

    const int m0 = blockIdx.y * TM;
    const int n0 = blockIdx.x * TN;

    // Async DMA staging: tile = 128 rows x 64 f16 = 1024 16B-chunks.
    // Thread t covers chunks {t + i*256}; chunk c -> row c>>3, 16B-col c&7.
    // 8 async-load instructions per wave per stage (4 A + 4 B).
    auto issue_stage = [&](int k0, int buf) {
#pragma unroll
        for (int i = 0; i < 4; ++i) {
            const int c   = t + i * 256;
            const int row = c >> 3;
            const int col = (c & 7) * 8;       // f16 units
            const _Float16* ga = Xh + (size_t)(m0 + row) * K + k0 + col;
            unsigned la = (unsigned)(unsigned long long)&As[buf][row * LDA + col];
            asm volatile("global_load_async_to_lds_b128 %0, %1, off"
                         :: "v"(la), "v"(ga) : "memory");
            const _Float16* gb = Bt + (size_t)(n0 + row) * K + k0 + col;
            unsigned lb = (unsigned)(unsigned long long)&Bs[buf][row * LDB + col];
            asm volatile("global_load_async_to_lds_b128 %0, %1, off"
                         :: "v"(lb), "v"(gb) : "memory");
        }
    };

    v8f acc[4][2];
#pragma unroll
    for (int mf = 0; mf < 4; ++mf)
#pragma unroll
        for (int nf = 0; nf < 2; ++nf) acc[mf][nf] = (v8f)(0.0f);

    const int NT = K / TK;
    issue_stage(0, 0);

    for (int kt = 0; kt < NT; ++kt) {
        const int cur = kt & 1;
        __syncthreads();                       // everyone done reading buf[cur^1]
        if (kt + 1 < NT) {
            issue_stage((kt + 1) * TK, cur ^ 1);
            // stage kt's 8 loads complete in-order; kt+1's 8 stay in flight
            asm volatile("s_wait_asynccnt 0x8" ::: "memory");
        } else {
            asm volatile("s_wait_asynccnt 0x0" ::: "memory");
        }
        __syncthreads();                       // stage kt visible to all waves

#pragma unroll
        for (int ks = 0; ks < 2; ++ks) {
            const int kk = ks * 32;
            Frag16 a[4], b[2];
            // A 16x32 f16: lanes<16 rows m, K{0..7,16..23}; lanes>=16 K{8..15,24..31}
#pragma unroll
            for (int mf = 0; mf < 4; ++mf) {
                const _Float16* ap = &As[cur][(waveM * 64 + mf * 16 + l) * LDA + kk];
                a[mf].u[0] = *(const uint4*)(ap + half * 8);
                a[mf].u[1] = *(const uint4*)(ap + 16 + half * 8);
            }
            // B 32x16 f16: lane holds one N column, lanes<16 K{0..7,16..23}; lanes>=16 K{8..15,24..31}
#pragma unroll
            for (int nf = 0; nf < 2; ++nf) {
                const _Float16* bp = &Bs[cur][(waveN * 32 + nf * 16 + l) * LDB + kk];
                b[nf].u[0] = *(const uint4*)(bp + half * 8);
                b[nf].u[1] = *(const uint4*)(bp + 16 + half * 8);
            }
#pragma unroll
            for (int mf = 0; mf < 4; ++mf)
#pragma unroll
                for (int nf = 0; nf < 2; ++nf)
                    acc[mf][nf] = __builtin_amdgcn_wmma_f32_16x16x32_f16(
                        false, a[mf].h, false, b[nf].h,
                        (short)0, acc[mf][nf], false, false);
        }
    }

    // Epilogue: C layout VGPR r, lane L -> (m = r + 8*half, n = L&15)
    const int nl = n0 + waveN * 32 + l;
#pragma unroll
    for (int nf = 0; nf < 2; ++nf) {
        const float bv = bias[nl + nf * 16];
#pragma unroll
        for (int mf = 0; mf < 4; ++mf) {
#pragma unroll
            for (int r = 0; r < 8; ++r) {
                const int mg = m0 + waveM * 64 + mf * 16 + r + half * 8;
                Y[(size_t)mg * N + nl + nf * 16] = acc[mf][nf][r] + bv;
            }
        }
    }
}

// ---------------------------------------------------------------------------
extern "C" void kernel_launch(void* const* d_in, const int* in_sizes, int n_in,
                              void* d_out, int out_size, void* d_ws, size_t ws_size,
                              hipStream_t stream) {
    const float* x      = (const float*)d_in[0];
    const float* scales = (const float*)d_in[1];
    const float* bias   = (const float*)d_in[2];
    const int*   W      = (const int*)d_in[3];
    const int*   ci     = (const int*)d_in[4];

    const int K      = in_sizes[4];            // col_indices length
    const int N      = in_sizes[2];            // bias length
    const int M      = in_sizes[0] / K;        // B*S
    const int GROUPS = in_sizes[1] / N;
    const int GS     = K / GROUPS;             // 128

    // workspace layout: [inv: K ints][Xh: M*K f16][Bt: N*K f16], 256B aligned
    char*     base = (char*)d_ws;
    int*      inv  = (int*)base;
    size_t    off  = ((size_t)K * sizeof(int) + 255) & ~(size_t)255;
    _Float16* Xh   = (_Float16*)(base + off);
    off += (((size_t)M * K * sizeof(_Float16)) + 255) & ~(size_t)255;
    _Float16* Bt   = (_Float16*)(base + off);

    k_invert<<<(K + 255) / 256, 256, 0, stream>>>(ci, inv, K);

    size_t nx = (size_t)M * K;
    k_convert<<<(unsigned)(nx / 8 / 256), 256, 0, stream>>>(x, Xh, nx);

    dim3 gD(N / 64, K / 64);
    k_dequant<<<gD, 256, 0, stream>>>(W, scales, inv, Bt, K, N, GS);

    dim3 gG(N / TN, M / TM);
    k_gemm<<<gG, 256, 0, stream>>>(Xh, Bt, bias, (float*)d_out, M, N, K);
}
